// GATHAConv_54262616817870
// MI455X (gfx1250) — compile-verified, matrix operand order checked by reading, not run on hardware
//
#include <hip/hip_runtime.h>
#include <hip/hip_bf16.h>

// ---------------- problem constants (from reference) ----------------
#define N_NODES 50000
#define E_EDGES 800000
#define IN_F    256
#define H_HEADS 4
#define F_OUT   64
#define HF      256          // H_HEADS * F_OUT
#define SLOPE   0.2f

typedef __attribute__((ext_vector_type(16))) __bf16 v16bf;
typedef __attribute__((ext_vector_type(8)))  float  v8f;

// ---------------- helpers ----------------
__device__ __forceinline__ float leakyf(float x) { return x > 0.f ? x : SLOPE * x; }

// float atomic max via int/uint trick (valid with init = -1e30f)
__device__ __forceinline__ void atomicMaxF(float* addr, float val) {
  if (val >= 0.f) {
    atomicMax((int*)addr, __float_as_int(val));
  } else {
    atomicMin((unsigned int*)addr, __float_as_uint(val));
  }
}

// ---------------- Kernel 1: h = feat @ W via bf16 WMMA ----------------
// Block = 256 threads = 8 waves. Each wave: 16 rows x all 256 cols.
// W K-slice (32 x 256) staged transposed into LDS as bf16, stride 40 (pad 8).
// B-fragment loads are software-pipelined one N-tile ahead so ds_load_b128s
// overlap the previous v_wmma instead of serializing on s_wait_dscnt 0.
#define WT_STRIDE 40

__global__ __launch_bounds__(256)
void gemm_h_kernel(const float* __restrict__ feat, const float* __restrict__ W,
                   float* __restrict__ h) {
  __shared__ __align__(16) __bf16 Wt[HF * WT_STRIDE];   // 20480 bytes

  const int t       = threadIdx.x;
  const int wave    = t >> 5;
  const int lane    = t & 31;
  const int rowBase = blockIdx.x * 128 + wave * 16;
  const bool rowValid = (rowBase < N_NODES);            // wave-uniform
  const int nloc = lane & 15;
  const int kg   = lane >> 4;

  v8f acc[16];
#pragma unroll
  for (int nt = 0; nt < 16; ++nt)
#pragma unroll
    for (int v = 0; v < 8; ++v) acc[nt][v] = 0.f;

  for (int ks = 0; ks < IN_F / 32; ++ks) {
    const int kbase = ks * 32;
    __syncthreads();
    // stage W[kbase..kbase+32) x [0..256) transposed: Wt[col][krel] (bf16)
#pragma unroll
    for (int kk = 0; kk < 32; ++kk)
      Wt[t * WT_STRIDE + kk] = (__bf16)W[(kbase + kk) * HF + t];
    __syncthreads();

    if (rowValid) {
      // ---- A fragment: row = rowBase+nloc; K mapping per ISA 16-bit A table
      const float* arow = feat + (size_t)(rowBase + nloc) * IN_F + kbase;
      float4 a0 = *(const float4*)(arow + 8 * kg);
      float4 a1 = *(const float4*)(arow + 8 * kg + 4);
      float4 a2 = *(const float4*)(arow + 16 + 8 * kg);
      float4 a3 = *(const float4*)(arow + 16 + 8 * kg + 4);
      float atmp[16] = {a0.x, a0.y, a0.z, a0.w, a1.x, a1.y, a1.z, a1.w,
                        a2.x, a2.y, a2.z, a2.w, a3.x, a3.y, a3.z, a3.w};
      v16bf afrag;
#pragma unroll
      for (int i = 0; i < 16; ++i) afrag[i] = (__bf16)atmp[i];

      // ---- 16 N-tiles, B fragments double-buffered from LDS
      const __bf16* wrow0 = &Wt[nloc * WT_STRIDE];
      uint4 b0 = *(const uint4*)(wrow0 + 8 * kg);        // K = 8kg .. 8kg+7
      uint4 b1 = *(const uint4*)(wrow0 + 16 + 8 * kg);   // K = 16+8kg ..
#pragma unroll
      for (int nt = 0; nt < 16; ++nt) {
        uint4 nb0, nb1;
        if (nt < 15) {
          const __bf16* wrow = &Wt[((nt + 1) * 16 + nloc) * WT_STRIDE];
          nb0 = *(const uint4*)(wrow + 8 * kg);
          nb1 = *(const uint4*)(wrow + 16 + 8 * kg);
        }
        const __bf16* p0 = (const __bf16*)&b0;
        const __bf16* p1 = (const __bf16*)&b1;
        v16bf bfrag;
#pragma unroll
        for (int i = 0; i < 8; ++i) { bfrag[i] = p0[i]; bfrag[8 + i] = p1[i]; }
        acc[nt] = __builtin_amdgcn_wmma_f32_16x16x32_bf16(
            false, afrag, false, bfrag, (short)0, acc[nt], false, false);
        if (nt < 15) { b0 = nb0; b1 = nb1; }
      }
    }
  }

  if (rowValid) {
    // C/D layout: VGPR v -> M = v + 8*kg ; lane nloc -> N
#pragma unroll
    for (int nt = 0; nt < 16; ++nt)
#pragma unroll
      for (int v = 0; v < 8; ++v)
        h[(size_t)(rowBase + v + 8 * kg) * HF + nt * 16 + nloc] = acc[nt][v];
  }
}

// ---------------- Kernel 2: init accumulators ----------------
__global__ void init_kernel(float* maxs, float* maxd, float* sums, float* sumd,
                            float* outdeg, float* indeg) {
  int i = blockIdx.x * blockDim.x + threadIdx.x;
  if (i < N_NODES * H_HEADS) {
    maxs[i] = -1e30f; maxd[i] = -1e30f; sums[i] = 0.f; sumd[i] = 0.f;
  }
  if (i < N_NODES) { outdeg[i] = 0.f; indeg[i] = 0.f; }
}

// ---------------- Kernel 3: per-(node,head) attention scores ----------------
__global__ void scores_kernel(const float* __restrict__ h,
                              const float* __restrict__ attn_l,
                              const float* __restrict__ attn_r,
                              const float* __restrict__ hop_attn_l,
                              float* __restrict__ el, float* __restrict__ er,
                              float* __restrict__ al) {
  int i = blockIdx.x * blockDim.x + threadIdx.x;
  if (i >= N_NODES * H_HEADS) return;
  int node = i >> 2, head = i & 3;
  const float* hp  = h + (size_t)node * HF + head * F_OUT;
  const float* alp = attn_l + head * F_OUT;
  const float* arp = attn_r + head * F_OUT;
  const float* hlp = hop_attn_l + head * F_OUT;
  float sl = 0.f, sr = 0.f, shl = 0.f;
#pragma unroll 8
  for (int f = 0; f < F_OUT; ++f) {
    float v = hp[f];
    sl += v * alp[f]; sr += v * arp[f]; shl += v * hlp[f];
  }
  el[i] = sl; er[i] = sr; al[i] = shl;
}

// ---------------- Kernel 4: e = leaky(el[src]+er[dst]); seg-max; degrees ----
__global__ void edge_e_kernel(const int* __restrict__ src, const int* __restrict__ dst,
                              const float* __restrict__ el, const float* __restrict__ er,
                              float* __restrict__ e_ws, float* maxs, float* maxd,
                              float* outdeg, float* indeg) {
  int i = blockIdx.x * blockDim.x + threadIdx.x;
  if (i >= E_EDGES * H_HEADS) return;
  int eid = i >> 2, head = i & 3;
  int s = src[eid], d = dst[eid];
  float ev = leakyf(el[s * 4 + head] + er[d * 4 + head]);
  e_ws[i] = ev;
  atomicMaxF(&maxs[s * 4 + head], ev);
  atomicMaxF(&maxd[d * 4 + head], ev);
  if (head == 0) { atomicAdd(&outdeg[s], 1.f); atomicAdd(&indeg[d], 1.f); }
}

// ---------------- Kernel 5: segment exp-sums ----------------
__global__ void edge_sum_kernel(const int* __restrict__ src, const int* __restrict__ dst,
                                const float* __restrict__ e_ws,
                                const float* __restrict__ maxs, const float* __restrict__ maxd,
                                float* sums, float* sumd) {
  int i = blockIdx.x * blockDim.x + threadIdx.x;
  if (i >= E_EDGES * H_HEADS) return;
  int eid = i >> 2, head = i & 3;
  int s = src[eid], d = dst[eid];
  float ev = e_ws[i];
  atomicAdd(&sums[s * 4 + head], __expf(ev - maxs[s * 4 + head]));
  atomicAdd(&sumd[d * 4 + head], __expf(ev - maxd[d * 4 + head]));
}

// ---------------- Kernel 6: mixed attention a = a_dst^s * a_src^(1-s) -------
__global__ void edge_a_kernel(const int* __restrict__ src, const int* __restrict__ dst,
                              const float* __restrict__ e_ws,
                              const float* __restrict__ maxs, const float* __restrict__ maxd,
                              const float* __restrict__ sums, const float* __restrict__ sumd,
                              const float* __restrict__ sigma, float* __restrict__ a_ws) {
  int i = blockIdx.x * blockDim.x + threadIdx.x;
  if (i >= E_EDGES * H_HEADS) return;
  int eid = i >> 2, head = i & 3;
  int s = src[eid], d = dst[eid];
  float ev = e_ws[i];
  float sgm = 1.f / (1.f + __expf(-sigma[0]));
  float ax = fmaxf(__expf(ev - maxs[s * 4 + head]) / sums[s * 4 + head], 1e-10f);
  float ay = fmaxf(__expf(ev - maxd[d * 4 + head]) / sumd[d * 4 + head], 1e-10f);
  a_ws[i] = __powf(ay, sgm) * __powf(ax, 1.f - sgm);
}

// ---------------- Kernel 7: degree norms ----------------
__global__ void norm_kernel(const float* __restrict__ outdeg, const float* __restrict__ indeg,
                            float* __restrict__ norm_out, float* __restrict__ norm_in) {
  int i = blockIdx.x * blockDim.x + threadIdx.x;
  if (i >= N_NODES) return;
  norm_out[i] = rsqrtf(fmaxf(outdeg[i], 1.f));   // out_deg^-0.5
  norm_in[i]  = sqrtf(fmaxf(indeg[i], 1.f));     // in_deg^+0.5
}

// ---------------- Kernel 8: hop aggregation (one block per edge) ------------
__global__ __launch_bounds__(256)
void agg_kernel(const int* __restrict__ src, const int* __restrict__ dst,
                const float* __restrict__ a_ws, const float* __restrict__ norm_out,
                const float* __restrict__ xprev, float* __restrict__ xnext) {
  int eid = blockIdx.x;
  int t = threadIdx.x;              // = head*64 + f
  int s = src[eid], d = dst[eid];   // uniform -> scalar loads
  int head = t >> 6;
  float coef = a_ws[eid * 4 + head] * norm_out[s];
  float v = xprev[(size_t)s * HF + t] * coef;
  atomicAdd(&xnext[(size_t)d * HF + t], v);
}

// ---------------- Kernel 9: scale by in-norm ----------------
__global__ void scale_kernel(float* __restrict__ x, const float* __restrict__ norm_in) {
  int i = blockIdx.x * blockDim.x + threadIdx.x;
  if (i >= N_NODES * HF) return;
  x[i] *= norm_in[i >> 8];
}

// ---------------- Kernel 10: hop-attention combine (wave per node,head) -----
__global__ __launch_bounds__(128)
void combine_kernel(const float* __restrict__ h, const float* __restrict__ x1,
                    const float* __restrict__ x2, const float* __restrict__ x3,
                    const float* __restrict__ al, const float* __restrict__ hop_attn_r,
                    float* __restrict__ out) {
  int n    = blockIdx.x;
  int head = threadIdx.x >> 5;      // 4 waves = 4 heads
  int lane = threadIdx.x & 31;
  const float* hs[4] = {h, x1, x2, x3};
  size_t base = (size_t)n * HF + head * F_OUT;
  const float* ar = hop_attn_r + head * F_OUT;
  float ar0 = ar[lane], ar1 = ar[lane + 32];
  float albase = al[n * 4 + head];

  float xv0[4], xv1[4], score[4];
#pragma unroll
  for (int k = 0; k < 4; ++k) {
    float v0 = hs[k][base + lane];
    float v1 = hs[k][base + lane + 32];
    xv0[k] = v0; xv1[k] = v1;
    float p = v0 * ar0 + v1 * ar1;
#pragma unroll
    for (int off = 16; off > 0; off >>= 1) p += __shfl_xor(p, off);
    score[k] = leakyf(albase + p);
  }
  float m = fmaxf(fmaxf(score[0], score[1]), fmaxf(score[2], score[3]));
  float ex[4], ssum = 0.f;
#pragma unroll
  for (int k = 0; k < 4; ++k) { ex[k] = __expf(score[k] - m); ssum += ex[k]; }
  float o0 = 0.f, o1 = 0.f;
#pragma unroll
  for (int k = 0; k < 4; ++k) {
    float w = ex[k] / ssum;
    o0 += xv0[k] * w; o1 += xv1[k] * w;
  }
  out[base + lane] = o0;
  out[base + lane + 32] = o1;
}

// ---------------- launcher ----------------
extern "C" void kernel_launch(void* const* d_in, const int* in_sizes, int n_in,
                              void* d_out, int out_size, void* d_ws, size_t ws_size,
                              hipStream_t stream) {
  (void)in_sizes; (void)n_in; (void)out_size; (void)ws_size;
  const float* feat       = (const float*)d_in[0];
  const int*   src        = (const int*)d_in[1];
  const int*   dst        = (const int*)d_in[2];
  const float* W          = (const float*)d_in[3];
  const float* attn_l     = (const float*)d_in[4];
  const float* attn_r     = (const float*)d_in[5];
  const float* hop_attn_l = (const float*)d_in[6];
  const float* hop_attn_r = (const float*)d_in[7];
  const float* sigma      = (const float*)d_in[8];

  // workspace carve-up (floats)
  float* ws = (float*)d_ws;
  size_t off = 0;
  float* h    = ws + off; off += (size_t)N_NODES * HF;
  float* x1   = ws + off; off += (size_t)N_NODES * HF;
  float* x2   = ws + off; off += (size_t)N_NODES * HF;
  float* x3   = ws + off; off += (size_t)N_NODES * HF;
  float* el   = ws + off; off += (size_t)N_NODES * H_HEADS;
  float* er   = ws + off; off += (size_t)N_NODES * H_HEADS;
  float* al   = ws + off; off += (size_t)N_NODES * H_HEADS;
  float* maxs = ws + off; off += (size_t)N_NODES * H_HEADS;
  float* maxd = ws + off; off += (size_t)N_NODES * H_HEADS;
  float* sums = ws + off; off += (size_t)N_NODES * H_HEADS;
  float* sumd = ws + off; off += (size_t)N_NODES * H_HEADS;
  float* outdeg   = ws + off; off += (size_t)N_NODES;
  float* indeg    = ws + off; off += (size_t)N_NODES;
  float* norm_out = ws + off; off += (size_t)N_NODES;
  float* norm_in  = ws + off; off += (size_t)N_NODES;
  float* e_ws = ws + off; off += (size_t)E_EDGES * H_HEADS;
  float* a_ws = ws + off; off += (size_t)E_EDGES * H_HEADS;

  // zero the three hop accumulators (contiguous)
  hipMemsetAsync(x1, 0, (size_t)3 * N_NODES * HF * sizeof(float), stream);

  // 1) FC layer via WMMA
  gemm_h_kernel<<<(N_NODES + 127) / 128, 256, 0, stream>>>(feat, W, h);

  // 2) init segment accumulators
  const int NH = N_NODES * H_HEADS;
  init_kernel<<<(NH + 255) / 256, 256, 0, stream>>>(maxs, maxd, sums, sumd, outdeg, indeg);

  // 3) node scores
  scores_kernel<<<(NH + 255) / 256, 256, 0, stream>>>(h, attn_l, attn_r, hop_attn_l, el, er, al);

  // 4-6) edge sweeps
  const int EH = E_EDGES * H_HEADS;
  edge_e_kernel<<<(EH + 255) / 256, 256, 0, stream>>>(src, dst, el, er, e_ws, maxs, maxd, outdeg, indeg);
  edge_sum_kernel<<<(EH + 255) / 256, 256, 0, stream>>>(src, dst, e_ws, maxs, maxd, sums, sumd);
  edge_a_kernel<<<(EH + 255) / 256, 256, 0, stream>>>(src, dst, e_ws, maxs, maxd, sums, sumd, sigma, a_ws);

  // 7) degree norms
  norm_kernel<<<(N_NODES + 255) / 256, 256, 0, stream>>>(outdeg, indeg, norm_out, norm_in);

  // 8-9) K=3 hops
  const int NE = N_NODES * HF;
  agg_kernel<<<E_EDGES, 256, 0, stream>>>(src, dst, a_ws, norm_out, h, x1);
  scale_kernel<<<(NE + 255) / 256, 256, 0, stream>>>(x1, norm_in);
  agg_kernel<<<E_EDGES, 256, 0, stream>>>(src, dst, a_ws, norm_out, x1, x2);
  scale_kernel<<<(NE + 255) / 256, 256, 0, stream>>>(x2, norm_in);
  agg_kernel<<<E_EDGES, 256, 0, stream>>>(src, dst, a_ws, norm_out, x2, x3);
  scale_kernel<<<(NE + 255) / 256, 256, 0, stream>>>(x3, norm_in);

  // 10) hop-attention combine -> output
  combine_kernel<<<N_NODES, 128, 0, stream>>>(h, x1, x2, x3, al, hop_attn_r, (float*)d_out);
}
